// STRecognizer_27092653703204
// MI455X (gfx1250) — matrix operand order repaired and checked by reading, not compiled, not run on hardware
//
#include <hip/hip_runtime.h>

// ---------------------------------------------------------------------------
// CDNA5 / gfx1250 implementation of the point-cloud decoder.
//  - KNN distance tiles:  V_WMMA_F32_16X16X4_F32  (K=3 coords pad to K=4)
//  - Channel GEMMs:       V_WMMA_F32_16X16X32_BF16 (f32 accumulate),
//    B fragments packed fragment-major -> 2x b128 loads per fragment,
//    A fragment reused across NT output tiles per wave.
//  - Top-3 selection: branchless compare-swap network (v_cndmask, no exec
//    divergence in the hot loop).
// ---------------------------------------------------------------------------

typedef __bf16 bf16x16 __attribute__((ext_vector_type(16)));
typedef float f32x8 __attribute__((ext_vector_type(8)));
typedef float f32x2 __attribute__((ext_vector_type(2)));
typedef unsigned int u32x4 __attribute__((ext_vector_type(4)));

#define EPS_LN 1e-5f
#define EPS_BN 1e-5f
#define BIGD 1.0e10f

__device__ __forceinline__ unsigned short f2bf(float f) {
  unsigned u = __builtin_bit_cast(unsigned, f);
  unsigned r = u + 0x7FFFu + ((u >> 16) & 1u);          // round-to-nearest-even
  if ((u & 0x7F800000u) == 0x7F800000u) r = u;          // Inf/NaN: truncate
  return (unsigned short)(r >> 16);
}

// Branchless insert of (x, ix) into sorted triple s0<=s1<=s2 (3 cmp + 10 sel).
__device__ __forceinline__ void insert3(float x, int ix,
                                        float& s0, float& s1, float& s2,
                                        int& i0, int& i1, int& i2) {
  bool c2 = x < s2;                 // evict current 3rd if candidate smaller
  float t2 = c2 ? x : s2;
  int j2 = c2 ? ix : i2;
  bool c1 = t2 < s1;                // compare-swap (s1, t2)
  float ns1 = c1 ? t2 : s1;
  float ns2 = c1 ? s1 : t2;
  int nj1 = c1 ? j2 : i1;
  int nj2 = c1 ? i1 : j2;
  bool c0 = ns1 < s0;               // compare-swap (s0, ns1)
  float fs0 = c0 ? ns1 : s0;
  float fs1 = c0 ? s0 : ns1;
  int fj0 = c0 ? nj1 : i0;
  int fj1 = c0 ? i0 : nj1;
  s0 = fs0; s1 = fs1; s2 = ns2;
  i0 = fj0; i1 = fj1; i2 = nj2;
}

// ---------------------------------------------------------------------------
// LayerNorm (one wave per row) -> bf16 activations
// ---------------------------------------------------------------------------
__global__ void __launch_bounds__(256)
ln_kernel(const float* __restrict__ X, const float* __restrict__ g,
          const float* __restrict__ b, unsigned short* __restrict__ out,
          int N, int C) {
  int row = (int)((blockIdx.x * blockDim.x + threadIdx.x) >> 5);
  int lane = threadIdx.x & 31;
  if (row >= N) return;
  const float* xr = X + (size_t)row * C;
  float s = 0.f, s2 = 0.f;
  for (int c = lane; c < C; c += 32) { float x = xr[c]; s += x; s2 += x * x; }
#pragma unroll
  for (int m = 16; m >= 1; m >>= 1) {
    s += __shfl_xor(s, m, 32);
    s2 += __shfl_xor(s2, m, 32);
  }
  float mean = s / (float)C;
  float var = fmaxf(s2 / (float)C - mean * mean, 0.f);
  float inv = rsqrtf(var + EPS_LN);
  unsigned short* orow = out + (size_t)row * C;
  for (int c = lane; c < C; c += 32) {
    float y = (xr[c] - mean) * inv * g[c] + b[c];
    orow[c] = f2bf(y);
  }
}

// ---------------------------------------------------------------------------
// Pack W[Cin,Cout] (f32) into fragment-major bf16 pairs:
//   Wq[(((k0/32)*2 + h)*Cout + col)*8 + v] = {W[k][col], W[k+1][col]},
//   k = k0 + 16h + 2v.  A lane's whole B fragment = 8 consecutive u32s.
// ---------------------------------------------------------------------------
__global__ void packw_kernel(const float* __restrict__ W, unsigned* __restrict__ Wq,
                             int Cin, int Cout) {
  int t = blockIdx.x * blockDim.x + threadIdx.x;
  int total = (Cin >> 1) * Cout;
  if (t >= total) return;
  int v = t & 7;
  int rest = t >> 3;
  int col = rest % Cout;
  int rest2 = rest / Cout;
  int hh = rest2 & 1;
  int ks = rest2 >> 1;
  int k = ks * 32 + hh * 16 + 2 * v;
  unsigned lo = f2bf(W[(size_t)k * Cout + col]);
  unsigned hi = f2bf(W[(size_t)(k + 1) * Cout + col]);
  Wq[t] = lo | (hi << 16);
}

// ---------------------------------------------------------------------------
// Y[M,Cout] = Xn(bf16)[M,Cin] @ W + bias (+ addin).  One wave computes a
// 16 x (NT*16) output strip: A fragment (2x b128) reused across NT WMMAs,
// each B fragment is 2x b128 from the fragment-major pack.
// ---------------------------------------------------------------------------
template <int NT>
__global__ void __launch_bounds__(256)
gemm_kernel(const unsigned short* __restrict__ Xn, const unsigned* __restrict__ Wq,
            const float* __restrict__ bias, const float* __restrict__ addin,
            int hasAdd, float* __restrict__ Y, int M, int Cin, int Cout) {
  int wid = (int)((blockIdx.x * blockDim.x + threadIdx.x) >> 5);
  int lane = threadIdx.x & 31;
  int ngroups = Cout / (16 * NT);
  int mtiles = (M + 15) >> 4;
  if (wid >= mtiles * ngroups) return;
  int mtile = wid / ngroups, ng = wid % ngroups;
  int h = lane >> 4, l16 = lane & 15;

  int row = mtile * 16 + l16;
  int rowc = row < M ? row : M - 1;
  const unsigned short* xr = Xn + (size_t)rowc * Cin;
  int col0 = ng * NT * 16 + l16;

  f32x8 acc[NT];
#pragma unroll
  for (int nt = 0; nt < NT; ++nt) acc[nt] = (f32x8){};

  int ksteps = Cin >> 5;
  for (int ks = 0; ks < ksteps; ++ks) {
    int k0 = ks << 5;
    // A 16x32 bf16 fragment: lane-half h holds K = k0+16h+0..15
    u32x4 a0 = *(const u32x4*)(xr + k0 + 16 * h);
    u32x4 a1 = *(const u32x4*)(xr + k0 + 16 * h + 8);
    bf16x16 a = __builtin_bit_cast(
        bf16x16, __builtin_shufflevector(a0, a1, 0, 1, 2, 3, 4, 5, 6, 7));
    const unsigned* wbase = Wq + ((size_t)(ks * 2 + h) * Cout + col0) * 8;
#pragma unroll
    for (int nt = 0; nt < NT; ++nt) {
      const u32x4* wp4 = (const u32x4*)(wbase + (size_t)nt * 16 * 8);
      u32x4 b0 = wp4[0];
      u32x4 b1 = wp4[1];
      bf16x16 bm = __builtin_bit_cast(
          bf16x16, __builtin_shufflevector(b0, b1, 0, 1, 2, 3, 4, 5, 6, 7));
      acc[nt] = __builtin_amdgcn_wmma_f32_16x16x32_bf16(false, a, false, bm,
                                                        (short)0, acc[nt],
                                                        false, false);
    }
  }
#pragma unroll
  for (int nt = 0; nt < NT; ++nt) {
#pragma unroll
    for (int r = 0; r < 8; ++r) {
      int orow = mtile * 16 + r + 8 * h;   // D layout: lane=col, VGPR r -> row
      if (orow < M) {
        int col = col0 + nt * 16;
        size_t o = (size_t)orow * Cout + col;
        float v = acc[nt][r] + bias[col];
        if (hasAdd) v += addin[o];
        Y[o] = v;
      }
    }
  }
}

// ---------------------------------------------------------------------------
// KNN (k=3) via f32 WMMA distance tiles. One wave handles 16 fine points,
// streams the coarse set in 16-column tiles, keeps per-lane running top-3
// (branchless), then merges 16 lane-candidates per row through LDS and emits
// normalized inverse-distance weights + indices.
// ---------------------------------------------------------------------------
__global__ void __launch_bounds__(256)
knn_kernel(const float* __restrict__ fxyz, const float* __restrict__ cxyz,
           const int* __restrict__ foff, const int* __restrict__ coff,
           int Nf, int Nc, int* __restrict__ oidx, float* __restrict__ ow) {
  __shared__ float cs[8][16][16][3];
  __shared__ int ci[8][16][16][3];
  int w = threadIdx.x >> 5;
  int lane = threadIdx.x & 31;
  int h = lane >> 4, l16 = lane & 15;
  int mbase = (int)(blockIdx.x * 8 + w) * 16;

  int foff0 = foff[0];
  int coff0 = coff[0];

  int m = mbase + l16;
  int mc = m < Nf ? m : Nf - 1;
  float fx = fxyz[3 * mc], fy = fxyz[3 * mc + 1], fz = fxyz[3 * mc + 2];
  float ff = fx * fx + fy * fy + fz * fz;

  // A row (M = l16): (-2fx, -2fy, -2fz, 1); f32 A 16x4: VGPR0 K=2h, VGPR1 K=2h+1
  f32x2 a;
  a[0] = h ? (-2.f * fz) : (-2.f * fx);
  a[1] = h ? 1.f : (-2.f * fy);

  int clo[8], chi[8];
#pragma unroll
  for (int r = 0; r < 8; ++r) {
    int rm = mbase + r + 8 * h;           // row handled by (this half, VGPR r)
    bool inB1 = (rm >= foff0);
    clo[r] = inB1 ? coff0 : 0;
    chi[r] = inB1 ? Nc : coff0;           // chi <= Nc also guards tail padding
  }

  float s0[8], s1[8], s2[8];
  int i0[8], i1[8], i2[8];
#pragma unroll
  for (int r = 0; r < 8; ++r) {
    s0[r] = s1[r] = s2[r] = 3.0e38f;
    i0[r] = i1[r] = i2[r] = 0;
  }

  for (int cb = 0; cb < Nc; cb += 16) {
    int n = cb + l16;
    int nc = n < Nc ? n : Nc - 1;
    float cx = cxyz[3 * nc], cy = cxyz[3 * nc + 1], cz = cxyz[3 * nc + 2];
    float cc = cx * cx + cy * cy + cz * cz;
    // B col (N = l16): (cx, cy, cz, |c|^2); VGPR0 K=2h, VGPR1 K=2h+1
    f32x2 bm;
    bm[0] = h ? cz : cx;
    bm[1] = h ? cc : cy;
    f32x8 d = {};
    d = __builtin_amdgcn_wmma_f32_16x16x4_f32(false, a, false, bm,
                                              (short)0, d, false, false);
#pragma unroll
    for (int r = 0; r < 8; ++r) {
      // score = |c|^2 - 2 f.c  (rank-equivalent to d2 within a row)
      bool valid = (n >= clo[r]) & (n < chi[r]);   // bitwise: no short-circuit
      float s = valid ? d[r] : BIGD;
      insert3(s, n, s0[r], s1[r], s2[r], i0[r], i1[r], i2[r]);
    }
  }

#pragma unroll
  for (int r = 0; r < 8; ++r) {
    int rowi = r + 8 * h;
    cs[w][rowi][l16][0] = s0[r]; ci[w][rowi][l16][0] = i0[r];
    cs[w][rowi][l16][1] = s1[r]; ci[w][rowi][l16][1] = i1[r];
    cs[w][rowi][l16][2] = s2[r]; ci[w][rowi][l16][2] = i2[r];
  }
  __syncthreads();

  if (h == 0) {                           // lanes 0..15: reduce row l16
    float b0 = 3.0e38f, b1v = 3.0e38f, b2v = 3.0e38f;
    int j0 = 0, j1 = 0, j2 = 0;
    for (int ln2 = 0; ln2 < 16; ++ln2) {
#pragma unroll
      for (int k = 0; k < 3; ++k) {
        insert3(cs[w][l16][ln2][k], ci[w][l16][ln2][k],
                b0, b1v, b2v, j0, j1, j2);
      }
    }
    if (m < Nf) {
      float d0 = sqrtf(fmaxf(b0 + ff, 0.f));
      float d1 = sqrtf(fmaxf(b1v + ff, 0.f));
      float d2 = sqrtf(fmaxf(b2v + ff, 0.f));
      float w0 = 1.f / (d0 + 1e-8f);
      float w1 = 1.f / (d1 + 1e-8f);
      float w2 = 1.f / (d2 + 1e-8f);
      float wsum = w0 + w1 + w2;
      oidx[3 * m + 0] = j0; oidx[3 * m + 1] = j1; oidx[3 * m + 2] = j2;
      ow[3 * m + 0] = w0 / wsum; ow[3 * m + 1] = w1 / wsum; ow[3 * m + 2] = w2 / wsum;
    }
  }
}

// ---------------------------------------------------------------------------
// interp: out[m,:] = sum_k w[m,k] * H2[idx[m,k],:]   (float4 vectorized)
// ---------------------------------------------------------------------------
__global__ void interp_kernel(const float* __restrict__ H2, const int* __restrict__ idx,
                              const float* __restrict__ wgt, float* __restrict__ out,
                              int Nf, int C) {
  int t = blockIdx.x * blockDim.x + threadIdx.x;
  int c4 = C >> 2;
  if (t >= Nf * c4) return;
  int mrow = t / c4, cc = (t % c4) * 4;
  int j0 = idx[3 * mrow], j1 = idx[3 * mrow + 1], j2 = idx[3 * mrow + 2];
  float w0 = wgt[3 * mrow], w1 = wgt[3 * mrow + 1], w2 = wgt[3 * mrow + 2];
  float4 v0 = *(const float4*)(H2 + (size_t)j0 * C + cc);
  float4 v1 = *(const float4*)(H2 + (size_t)j1 * C + cc);
  float4 v2 = *(const float4*)(H2 + (size_t)j2 * C + cc);
  float4 o;
  o.x = w0 * v0.x + w1 * v1.x + w2 * v2.x;
  o.y = w0 * v0.y + w1 * v1.y + w2 * v2.y;
  o.z = w0 * v0.z + w1 * v1.z + w2 * v2.z;
  o.w = w0 * v0.w + w1 * v1.w + w2 * v2.w;
  *(float4*)(out + (size_t)mrow * C + cc) = o;
}

// ---------------------------------------------------------------------------
// Head: x = relu(bn(F@W1+b1)) @ W2 + b2     (C=48, one point per thread)
// ---------------------------------------------------------------------------
__global__ void __launch_bounds__(256)
head_kernel(const float* __restrict__ F, const float* __restrict__ W1,
            const float* __restrict__ b1, const float* __restrict__ bng,
            const float* __restrict__ bnb, const float* __restrict__ bnm,
            const float* __restrict__ bnv, const float* __restrict__ W2,
            const float* __restrict__ b2, float* __restrict__ out, int N) {
  __shared__ float Ws[48 * 48], W2s[48], scl[48], sft[48], b1s[48];
  for (int t = threadIdx.x; t < 48 * 48; t += blockDim.x) Ws[t] = W1[t];
  if (threadIdx.x < 48) {
    int j = threadIdx.x;
    W2s[j] = W2[j];
    float sc = bng[j] * rsqrtf(bnv[j] + EPS_BN);
    scl[j] = sc;
    sft[j] = bnb[j] - bnm[j] * sc;
    b1s[j] = b1[j];
  }
  __syncthreads();
  int m = blockIdx.x * blockDim.x + threadIdx.x;
  if (m >= N) return;
  const float* fr = F + (size_t)m * 48;
  float acc[48];
#pragma unroll
  for (int j = 0; j < 48; ++j) acc[j] = 0.f;
  for (int c = 0; c < 48; ++c) {
    float x = fr[c];
#pragma unroll
    for (int j = 0; j < 48; ++j) acc[j] += x * Ws[c * 48 + j];
  }
  float o = 0.f;
#pragma unroll
  for (int j = 0; j < 48; ++j) {
    float y = acc[j] + b1s[j];
    y = fmaxf(y * scl[j] + sft[j], 0.f);
    o += y * W2s[j];
  }
  out[m] = o + b2[0];
}

// ---------------------------------------------------------------------------
// Host orchestration
// ---------------------------------------------------------------------------
static inline int idivu(long long a, long long b) { return (int)((a + b - 1) / b); }

static void launch_gemm(const unsigned short* Xn, const unsigned* Wq,
                        const float* bias, const float* addin, int hasAdd,
                        float* Y, int M, int Cin, int Cout, hipStream_t stream) {
  if (Cout % 96 == 0) {
    long long waves = (long long)((M + 15) / 16) * (Cout / 96);
    gemm_kernel<6><<<idivu(waves, 8), 256, 0, stream>>>(Xn, Wq, bias, addin,
                                                        hasAdd, Y, M, Cin, Cout);
  } else {
    long long waves = (long long)((M + 15) / 16) * (Cout / 48);
    gemm_kernel<3><<<idivu(waves, 8), 256, 0, stream>>>(Xn, Wq, bias, addin,
                                                        hasAdd, Y, M, Cin, Cout);
  }
}

extern "C" void kernel_launch(void* const* d_in, const int* in_sizes, int n_in,
                              void* d_out, int out_size, void* d_ws, size_t ws_size,
                              hipStream_t stream) {
  (void)in_sizes; (void)n_in; (void)out_size; (void)ws_size;
  static const int CHN[5] = {48, 96, 192, 384, 768};
  static const int NSZ[5] = {160000, 40000, 10000, 2500, 640};

  const float* feats4 = (const float*)d_in[0];
  const float* xyz[5];
  const int* off[5];
  for (int l = 0; l < 5; ++l) {
    xyz[l] = (const float*)d_in[1 + 2 * l];
    off[l] = (const int*)d_in[2 + 2 * l];
  }
  const float* skip[4];
  for (int l = 0; l < 4; ++l) skip[l] = (const float*)d_in[11 + l];

  // params flattened (jax tree order, dict keys sorted): conf then stages[0..3]
  // conf: W1, W2, b1, b2, bn_b, bn_g, bn_mean, bn_var
  const float* cW1 = (const float*)d_in[15];
  const float* cW2 = (const float*)d_in[16];
  const float* cb1 = (const float*)d_in[17];
  const float* cb2 = (const float*)d_in[18];
  const float* bnb = (const float*)d_in[19];
  const float* bng = (const float*)d_in[20];
  const float* bnm = (const float*)d_in[21];
  const float* bnv = (const float*)d_in[22];
  // stage i at 23+8i: W1, W2, b1, b2, ln1_b, ln1_g, ln2_b, ln2_g
  const float *sW1[4], *sW2[4], *sb1[4], *sb2[4], *ln1b[4], *ln1g[4], *ln2b[4], *ln2g[4];
  for (int i = 0; i < 4; ++i) {
    int base = 23 + 8 * i;
    sW1[i] = (const float*)d_in[base + 0];
    sW2[i] = (const float*)d_in[base + 1];
    sb1[i] = (const float*)d_in[base + 2];
    sb2[i] = (const float*)d_in[base + 3];
    ln1b[i] = (const float*)d_in[base + 4];
    ln1g[i] = (const float*)d_in[base + 5];
    ln2b[i] = (const float*)d_in[base + 6];
    ln2g[i] = (const float*)d_in[base + 7];
  }

  // workspace carve (bump allocator, 256B aligned)
  char* p = (char*)d_ws;
  auto carve = [&](size_t bytes) -> void* {
    void* r = (void*)p;
    p += (bytes + 255) & ~(size_t)255;
    return r;
  };
  float* bufA = (float*)carve((size_t)160000 * 48 * 4);
  float* bufB = (float*)carve((size_t)160000 * 48 * 4);
  float* H2 = (float*)carve((size_t)40000 * 48 * 4);          // max Nc*cout
  unsigned short* Xn = (unsigned short*)carve((size_t)160000 * 48 * 2);
  unsigned* Wq = (unsigned*)carve((size_t)147456 * 4);        // max (Cin/2)*Cout
  int* kidx = (int*)carve((size_t)160000 * 3 * 4);
  float* kw = (float*)carve((size_t)160000 * 3 * 4);

  const float* feats_cur = feats4;
  for (int i = 0; i < 4; ++i) {
    int cl = 4 - i, fl = 3 - i;
    int cin = CHN[cl], cout = CHN[fl];
    int Nc = NSZ[cl], Nf = NSZ[fl];
    float* outb = (i & 1) ? bufB : bufA;

    // 1) h2 = LN(feats_coarse) @ W2 + b2
    ln_kernel<<<idivu(Nc, 8), 256, 0, stream>>>(feats_cur, ln2g[i], ln2b[i], Xn, Nc, cin);
    packw_kernel<<<idivu((long long)(cin / 2) * cout, 256), 256, 0, stream>>>(sW2[i], Wq, cin, cout);
    launch_gemm(Xn, Wq, sb2[i], nullptr, 0, H2, Nc, cin, cout, stream);

    // 2) 3-NN inverse-distance interpolation coarse -> fine
    knn_kernel<<<idivu(Nf, 128), 256, 0, stream>>>(xyz[fl], xyz[cl], off[fl], off[cl],
                                                   Nf, Nc, kidx, kw);
    interp_kernel<<<idivu((long long)Nf * (cout / 4), 256), 256, 0, stream>>>(H2, kidx, kw,
                                                                              outb, Nf, cout);

    // 3) feats = LN(skip) @ W1 + b1 + interp
    ln_kernel<<<idivu(Nf, 8), 256, 0, stream>>>(skip[fl], ln1g[i], ln1b[i], Xn, Nf, cout);
    packw_kernel<<<idivu((long long)(cout / 2) * cout, 256), 256, 0, stream>>>(sW1[i], Wq, cout, cout);
    launch_gemm(Xn, Wq, sb1[i], outb, 1, outb, Nf, cout, cout, stream);

    feats_cur = outb;
  }

  head_kernel<<<idivu(160000, 256), 256, 0, stream>>>(feats_cur, cW1, cb1, bng, bnb, bnm, bnv,
                                                      cW2, cb2, (float*)d_out, 160000);
}
